// Attention_29145648071355
// MI455X (gfx1250) — matrix operand order repaired and checked by reading, not compiled, not run on hardware
//
#include <hip/hip_runtime.h>
#include <hip/hip_bf16.h>

// ---------------------------------------------------------------------------
// nGPT causal attention for MI455X (gfx1250)
//   - all matmuls via v_wmma_f32_16x16x32_f16
//   - attention K/V tiles staged to LDS with the Tensor Data Mover (TDM),
//     double buffered, s_wait_tensorcnt + workgroup barriers
// ---------------------------------------------------------------------------

typedef __attribute__((ext_vector_type(16))) _Float16 v16h;
typedef __attribute__((ext_vector_type(8)))  _Float16 v8h;
typedef __attribute__((ext_vector_type(8)))  float    v8f;
typedef __attribute__((ext_vector_type(4)))  unsigned tdm_v4u;
typedef __attribute__((ext_vector_type(8)))  int      tdm_v8i;
typedef __attribute__((ext_vector_type(4)))  int      tdm_v4i;

#define DIM       2048
#define HEADS     16
#define DIM_HEAD  128
#define SEQ       2048
#define BATCH     2
#define MTOT      (BATCH * SEQ)     // 4096
#define EPS       1e-12f
#define ATTN_SCALE 11.313708498984761f  // sqrt(128)

__device__ __forceinline__ v8f wmma16(v16h a, v16h b, v8f c) {
  return __builtin_amdgcn_wmma_f32_16x16x32_f16(false, a, false, b, (short)0, c,
                                                false, false);
}

// A operand (16x32 f16): lane l -> row = l&15; K chunks [8h,8h+8) and [16+8h,16+8h+8)
__device__ __forceinline__ v16h load_A16(const _Float16* __restrict__ base, int stride) {
  const int l = threadIdx.x & 31;
  const int row = l & 15;
  const int h = l >> 4;
  const _Float16* p = base + (size_t)row * stride + 8 * h;
  v8h lo = *(const v8h*)(p);
  v8h hi = *(const v8h*)(p + 16);
  return __builtin_shufflevector(lo, hi, 0, 1, 2, 3, 4, 5, 6, 7,
                                 8, 9, 10, 11, 12, 13, 14, 15);
}

// B operand (32x16 f16) from B^T stored row-major [N, K]:
// lane l -> N = l&15 (row of B^T); K chunks [8h,8h+8) and [16+8h,16+8h+8)
__device__ __forceinline__ v16h load_B16(const _Float16* __restrict__ base, int stride) {
  const int l = threadIdx.x & 31;
  const int row = l & 15;
  const int h = l >> 4;
  const _Float16* p = base + (size_t)row * stride + 8 * h;
  v8h lo = *(const v8h*)(p);
  v8h hi = *(const v8h*)(p + 16);
  return __builtin_shufflevector(lo, hi, 0, 1, 2, 3, 4, 5, 6, 7,
                                 8, 9, 10, 11, 12, 13, 14, 15);
}

// ---------------------------------------------------------------------------
// TDM: async 2D tile load (fp16 elements) global -> LDS.
// Descriptor packing per CDNA5 ISA (D# groups 0/1).
// ---------------------------------------------------------------------------
__device__ __forceinline__ void tdm_load_2d_f16(unsigned lds_off, unsigned long long gaddr,
                                                unsigned tensor_d0, unsigned tensor_d1,
                                                unsigned tile_d0, unsigned tile_d1,
                                                unsigned stride0_elems) {
  tdm_v4u g0;
  g0.x = 1u;                                            // count=1, is_restore=0
  g0.y = lds_off;                                       // lds_addr (bytes)
  g0.z = (unsigned)(gaddr & 0xffffffffull);             // global_addr[31:0]
  g0.w = (unsigned)((gaddr >> 32) & 0x01ffffffull)      // global_addr[56:32]
         | (2u << 30);                                  // type=2 ("image")
  tdm_v8i g1;
  g1[0] = (int)(1u << 16);                              // data_size=1 -> 2 bytes
  g1[1] = (int)((tensor_d0 & 0xffffu) << 16);           // tensor_dim0[15:0] @ bits63:48
  g1[2] = (int)((tensor_d0 >> 16) | ((tensor_d1 & 0xffffu) << 16));
  g1[3] = (int)((tensor_d1 >> 16) | (tile_d0 << 16));   // tile_dim0 @ bits127:112
  g1[4] = (int)(tile_d1);                               // tile_dim1 @ bits143:128
  g1[5] = (int)(stride0_elems);                         // tensor_dim0_stride[31:0]
  g1[6] = 0;
  g1[7] = 0;
  tdm_v4i z4 = {0, 0, 0, 0};
#if defined(__clang_major__) && (__clang_major__ >= 23)
  tdm_v8i z8 = {0, 0, 0, 0, 0, 0, 0, 0};
  __builtin_amdgcn_tensor_load_to_lds(g0, g1, z4, z4, z8, 0);
#else
  __builtin_amdgcn_tensor_load_to_lds(g0, g1, z4, z4, 0);
#endif
}

// ---------------------------------------------------------------------------
// Prep kernels
// ---------------------------------------------------------------------------

__global__ __launch_bounds__(256) void cvt_f16_kernel(const float* __restrict__ X,
                                                      _Float16* __restrict__ O, int total) {
  int i = blockIdx.x * 256 + threadIdx.x;
  if (i < total) O[i] = (_Float16)X[i];
}

__global__ __launch_bounds__(256) void rownorm_f16_kernel(const float* __restrict__ W,
                                                          _Float16* __restrict__ O, int K) {
  __shared__ float sred[256];
  const int r = blockIdx.x, t = threadIdx.x;
  const float* row = W + (size_t)r * K;
  float s = 0.f;
  for (int i = t; i < K; i += 256) { float w = row[i]; s += w * w; }
  sred[t] = s; __syncthreads();
  for (int off = 128; off > 0; off >>= 1) {
    if (t < off) sred[t] += sred[t + off];
    __syncthreads();
  }
  const float inv = 1.0f / fmaxf(sqrtf(sred[0]), EPS);
  for (int i = t; i < K; i += 256) O[(size_t)r * K + i] = (_Float16)(row[i] * inv);
}

__global__ __launch_bounds__(256) void colsq_kernel(const float* __restrict__ W,
                                                    float* __restrict__ colsq, int D, int E) {
  int e = blockIdx.x * 256 + threadIdx.x;
  if (e >= E) return;
  float s = 0.f;
  for (int d = 0; d < D; ++d) { float w = W[(size_t)d * E + e]; s += w * w; }
  colsq[e] = s;
}

__global__ __launch_bounds__(256) void colscale_f16_kernel(const float* __restrict__ W,
                                                           const float* __restrict__ colsq,
                                                           _Float16* __restrict__ O, int total) {
  int i = blockIdx.x * 256 + threadIdx.x;
  if (i >= total) return;
  int e = i & (DIM - 1);
  O[i] = (_Float16)(W[i] / fmaxf(sqrtf(colsq[e]), EPS));
}

// ---------------------------------------------------------------------------
// WMMA GEMM: C[M,N] = A[M,K] * Bw[N,K]^T   (A,Bw fp16 row-major)
// ---------------------------------------------------------------------------

template <typename OutT>
__global__ __launch_bounds__(256) void gemm_wmma_kernel(const _Float16* __restrict__ A,
                                                        const _Float16* __restrict__ Bw,
                                                        OutT* __restrict__ C,
                                                        int M, int N, int K) {
  const int lane = threadIdx.x & 31;
  const int wave = threadIdx.x >> 5;
  const int m0 = blockIdx.x * 128 + wave * 16;
  const int n0 = blockIdx.y * 128;
  v8f acc[8] = {};
  for (int k0 = 0; k0 < K; k0 += 32) {
    v16h a = load_A16(A + (size_t)m0 * K + k0, K);
#pragma unroll
    for (int t = 0; t < 8; ++t) {
      v16h b = load_B16(Bw + (size_t)(n0 + 16 * t) * K + k0, K);
      acc[t] = wmma16(a, b, acc[t]);
    }
  }
  const int cr = lane & 15, chh = lane >> 4;
#pragma unroll
  for (int t = 0; t < 8; ++t) {
#pragma unroll
    for (int r = 0; r < 8; ++r) {
      int row = m0 + r + 8 * chh;
      int col = n0 + 16 * t + cr;
      C[(size_t)row * N + col] = (OutT)acc[t][r];
    }
  }
}

// ---------------------------------------------------------------------------
// Per (b,h,n): l2norm q,k over head dim, fold qk_scale*DIM*sqrt(Dh) into q,
// scatter v into per-head transposed Vt[(b,h)][d][n].
// ---------------------------------------------------------------------------

__global__ __launch_bounds__(128) void qkv_norm_kernel(_Float16* __restrict__ q,
                                                       _Float16* __restrict__ k,
                                                       const _Float16* __restrict__ v,
                                                       _Float16* __restrict__ vt,
                                                       const float* __restrict__ qk_scale) {
  const int bid = blockIdx.x;
  const int h = bid & (HEADS - 1);
  const int n = (bid >> 4) & (SEQ - 1);
  const int b = bid >> 15;
  const int d = threadIdx.x;
  const int e = h * DIM_HEAD + d;
  const size_t idx = ((size_t)b * SEQ + n) * DIM + e;

  float qv = (float)q[idx], kv = (float)k[idx], vv = (float)v[idx];
  __shared__ float s[256];
  s[d] = qv * qv; s[128 + d] = kv * kv;
  __syncthreads();
  for (int off = 64; off > 0; off >>= 1) {
    if (d < off) { s[d] += s[d + off]; s[128 + d] += s[128 + d + off]; }
    __syncthreads();
  }
  float qn = qv / fmaxf(sqrtf(s[0]), EPS) * (qk_scale[e] * (float)DIM * ATTN_SCALE);
  float kn = kv / fmaxf(sqrtf(s[128]), EPS);
  q[idx] = (_Float16)qn;
  k[idx] = (_Float16)kn;
  vt[((size_t)(b * HEADS + h) * DIM_HEAD + d) * SEQ + n] = (_Float16)vv;
}

// ---------------------------------------------------------------------------
// Flash-style causal attention.
// Workgroup = 128 threads = 4 waves = 4 consecutive 16-query tiles of one (b,h).
// Shared j-loop: TDM double-buffers the K (32x128) and Vt (128x32) tiles in LDS.
// Scores computed transposed (A=K, B=Q -> St[j,i]) so softmax stats are per-lane
// and exp(St) registers map directly onto the A operand of P for O = P*V.
// grid = (SEQ/64, HEADS, BATCH), block = 128
// ---------------------------------------------------------------------------

__global__ __launch_bounds__(128) void attn_kernel(const _Float16* __restrict__ qn,
                                                   const _Float16* __restrict__ kn,
                                                   const _Float16* __restrict__ vt,
                                                   _Float16* __restrict__ O) {
  __shared__ alignas(64) _Float16 kbuf[2][32 * 128];  // [j 32][d 128]
  __shared__ alignas(64) _Float16 vbuf[2][128 * 32];  // [d 128][j 32]

  const int lane = threadIdx.x & 31;
  const int wave = threadIdx.x >> 5;
  const int wg_i0 = blockIdx.x * 64;
  const int i0 = wg_i0 + wave * 16;
  const int h = blockIdx.y;
  const int b = blockIdx.z;
  const int cr = lane & 15, chh = lane >> 4;

  const _Float16* qbase = qn + ((size_t)b * SEQ + i0) * DIM + h * DIM_HEAD;
  const _Float16* kbase = kn + (size_t)b * SEQ * DIM + h * DIM_HEAD;
  const _Float16* vbase = vt + (size_t)(b * HEADS + h) * DIM_HEAD * SEQ;

  v16h bq[4];
#pragma unroll
  for (int c = 0; c < 4; ++c) bq[c] = load_B16(qbase + 32 * c, DIM);

  v8f acc[8] = {};
  float m_run = -1e30f, l_run = 0.0f;
  const int iq = i0 + cr;  // this lane's query index (St column)

  const int t_last = (wg_i0 + 48) >> 5;  // shared loop bound (wave 3's range)

  // prefetch tile 0 into buffer 0 (issued once per workgroup by wave 0)
  if (wave == 0) {
    tdm_load_2d_f16((unsigned)(unsigned long long)(unsigned long long*)&kbuf[0][0],
                    (unsigned long long)(size_t)(kbase) * 0 +
                        (unsigned long long)(size_t)(const void*)kbase,
                    DIM, MTOT, DIM_HEAD, 32, DIM);
    tdm_load_2d_f16((unsigned)(unsigned long long)(unsigned long long*)&vbuf[0][0],
                    (unsigned long long)(size_t)(const void*)vbase,
                    SEQ, BATCH * HEADS * DIM_HEAD, 32, DIM_HEAD, SEQ);
  }

  for (int t = 0; t <= t_last; ++t) {
    const int j0 = t << 5;
    const int p = t & 1;
    if (wave == 0) __builtin_amdgcn_s_wait_tensorcnt(0);  // tile t landed
    __syncthreads();  // publish buf[p]; everyone done reading buf[1-p]
    if (wave == 0 && t < t_last) {
      const int jn = (t + 1) << 5;
      tdm_load_2d_f16((unsigned)(unsigned long long)(unsigned long long*)&kbuf[1 - p][0],
                      (unsigned long long)(size_t)(const void*)(kbase + (size_t)jn * DIM),
                      DIM, MTOT, DIM_HEAD, 32, DIM);
      tdm_load_2d_f16((unsigned)(unsigned long long)(unsigned long long*)&vbuf[1 - p][0],
                      (unsigned long long)(size_t)(const void*)(vbase + jn),
                      SEQ, BATCH * HEADS * DIM_HEAD, 32, DIM_HEAD, SEQ);
    }

    if (j0 <= i0) {  // wave-uniform causal participation
      const _Float16* kc = &kbuf[p][0];
      const _Float16* vc = &vbuf[p][0];
      v8f s0 = {}, s1 = {};
#pragma unroll
      for (int c = 0; c < 4; ++c) {
        v16h ak0 = load_A16(kc + 32 * c, DIM_HEAD);
        s0 = wmma16(ak0, bq[c], s0);
        v16h ak1 = load_A16(kc + 16 * DIM_HEAD + 32 * c, DIM_HEAD);
        s1 = wmma16(ak1, bq[c], s1);
      }

      float p0[8], p1[8];
      float mloc = -1e30f;
      if (j0 + 31 > i0) {  // diagonal block: per-element causal mask (once per wave)
#pragma unroll
        for (int r = 0; r < 8; ++r) {
          int ja = j0 + r + 8 * chh;
          int jb = j0 + 16 + r + 8 * chh;
          float sa = (ja <= iq) ? s0[r] : -1e30f;
          float sb = (jb <= iq) ? s1[r] : -1e30f;
          p0[r] = sa; p1[r] = sb;
          mloc = fmaxf(mloc, fmaxf(sa, sb));
        }
      } else {  // fully unmasked main path
#pragma unroll
        for (int r = 0; r < 8; ++r) {
          p0[r] = s0[r]; p1[r] = s1[r];
          mloc = fmaxf(mloc, fmaxf(p0[r], p1[r]));
        }
      }
      mloc = fmaxf(mloc, __shfl_xor(mloc, 16, 32));
      float m_new = fmaxf(m_run, mloc);
      float corr = __expf(m_run - m_new);

      float ssum = 0.f;
      v16h ap;  // P as A operand: [0,8)=St0 rows, [8,16)=St1 rows (in-lane)
#pragma unroll
      for (int r = 0; r < 8; ++r) {
        float ea = __expf(p0[r] - m_new);
        float eb = __expf(p1[r] - m_new);
        ssum += ea + eb;
        ap[r] = (_Float16)ea;
        ap[r + 8] = (_Float16)eb;
      }
      ssum += __shfl_xor(ssum, 16, 32);
      l_run = l_run * corr + ssum;

      if (__any(m_new > m_run)) {  // rescale O only when the max actually moved
#pragma unroll
        for (int r = 0; r < 8; ++r) {
          float c_r = __shfl(corr, r + 8 * chh, 32);
#pragma unroll
          for (int t2 = 0; t2 < 8; ++t2) acc[t2][r] *= c_r;
        }
      }
      m_run = m_new;

#pragma unroll
      for (int t2 = 0; t2 < 8; ++t2) {
        v16h bv = load_B16(vc + (size_t)(16 * t2) * 32, 32);
        acc[t2] = wmma16(ap, bv, acc[t2]);
      }
    }
  }

  // finalize: divide each row i by its softmax denom, store fp16 [b, n, h*128+d]
#pragma unroll
  for (int r = 0; r < 8; ++r) {
    float l_r = __shfl(l_run, r + 8 * chh, 32);
    float inv = 1.0f / l_r;
    int row = i0 + r + 8 * chh;
#pragma unroll
    for (int t = 0; t < 8; ++t) {
      O[((size_t)b * SEQ + row) * DIM + h * DIM_HEAD + 16 * t + cr] =
          (_Float16)(acc[t][r] * inv);
    }
  }
}

// ---------------------------------------------------------------------------
// Host launcher
// ---------------------------------------------------------------------------

extern "C" void kernel_launch(void* const* d_in, const int* in_sizes, int n_in,
                              void* d_out, int out_size, void* d_ws, size_t ws_size,
                              hipStream_t stream) {
  const float* x    = (const float*)d_in[0];
  const float* Wq   = (const float*)d_in[1];
  const float* Wk   = (const float*)d_in[2];
  const float* Wv   = (const float*)d_in[3];
  const float* Wout = (const float*)d_in[4];
  const float* qk   = (const float*)d_in[5];
  float* out        = (float*)d_out;

  char* ws = (char*)d_ws;
  const size_t H_X = (size_t)MTOT * DIM;   // 8388608 halfs
  const size_t H_W = (size_t)DIM * DIM;    // 4194304 halfs
  _Float16* x16  = (_Float16*)(ws);                         // 16 MB (reused as O16)
  _Float16* wq16 = (_Float16*)(ws + 16777216);              // 8 MB
  _Float16* wk16 = (_Float16*)(ws + 16777216 + 8388608);
  _Float16* wv16 = (_Float16*)(ws + 16777216 + 2 * 8388608);
  _Float16* wo16 = (_Float16*)(ws + 16777216 + 3 * 8388608);
  _Float16* q16  = (_Float16*)(ws + 16777216 + 4 * 8388608);           // 16 MB
  _Float16* k16  = (_Float16*)(ws + 16777216 + 4 * 8388608 + 16777216);
  _Float16* v16  = (_Float16*)(ws + 16777216 + 4 * 8388608 + 2 * 16777216);
  _Float16* vt16 = (_Float16*)(ws + 16777216 + 4 * 8388608 + 3 * 16777216);
  float*    csq  = (float*)   (ws + 16777216 + 4 * 8388608 + 4 * 16777216);
  _Float16* o16  = x16;  // reuse x16 region after the QKV GEMMs

  // 1) fp16 conversions / weight normalization
  cvt_f16_kernel<<<(int)(H_X / 256), 256, 0, stream>>>(x, x16, (int)H_X);
  rownorm_f16_kernel<<<DIM, 256, 0, stream>>>(Wq, wq16, DIM);
  rownorm_f16_kernel<<<DIM, 256, 0, stream>>>(Wk, wk16, DIM);
  rownorm_f16_kernel<<<DIM, 256, 0, stream>>>(Wv, wv16, DIM);
  colsq_kernel<<<DIM / 256, 256, 0, stream>>>(Wout, csq, DIM, DIM);
  colscale_f16_kernel<<<(int)(H_W / 256), 256, 0, stream>>>(Wout, csq, wo16, (int)H_W);

  // 2) QKV projections (WMMA)
  dim3 ggrid(MTOT / 128, DIM / 128);
  gemm_wmma_kernel<_Float16><<<ggrid, 256, 0, stream>>>(x16, wq16, q16, MTOT, DIM, DIM);
  gemm_wmma_kernel<_Float16><<<ggrid, 256, 0, stream>>>(x16, wk16, k16, MTOT, DIM, DIM);
  gemm_wmma_kernel<_Float16><<<ggrid, 256, 0, stream>>>(x16, wv16, v16, MTOT, DIM, DIM);

  // 3) per-head l2norm + qk scale + V transpose
  qkv_norm_kernel<<<BATCH * SEQ * HEADS, 128, 0, stream>>>(q16, k16, v16, vt16, qk);

  // 4) causal flash attention (WMMA + TDM double-buffered LDS staging)
  dim3 agrid(SEQ / 64, HEADS, BATCH);
  attn_kernel<<<agrid, 128, 0, stream>>>(q16, k16, vt16, o16);

  // 5) output projection (WMMA, f32 out)
  gemm_wmma_kernel<float><<<ggrid, 256, 0, stream>>>(o16, wo16, out, MTOT, DIM, DIM);
}
